// ASGLoss_67954972557837
// MI455X (gfx1250) — matrix operand order repaired and checked by reading, not compiled, not run on hardware
//
#include <hip/hip_runtime.h>
#include <math.h>

typedef _Float16 v16h  __attribute__((ext_vector_type(16)));
typedef _Float16 half8 __attribute__((ext_vector_type(8)));
typedef _Float16 half4 __attribute__((ext_vector_type(4)));
typedef float    v8f   __attribute__((ext_vector_type(8)));

#define NEGV -1e30f
constexpr int Bsz = 64;   // batch
constexpr int Lsz = 64;   // labels
constexpr int Ssz = 256;  // target length

// Raw-hardware transcendentals: args are finite, never denormal/inf/NaN in our use,
// so skip the OCML guard code (ldexp prescale, cndmask fixups, fma ln2 correction).
__device__ __forceinline__ float fast_exp(float x) {   // exp(x), x <= ~0, underflow->0 ok
  return __builtin_amdgcn_exp2f(x * 1.4426950408889634f);
}
__device__ __forceinline__ float fast_log(float x) {   // log(x), x in [~0.9, ~70]
  return __builtin_amdgcn_logf(x) * 0.6931471805599453f;
}

// A-fragment (16x32 f16) per CDNA5 ISA layout:
//   lanes 0-15  (grp=0), M=lane:    elems 0..7 = K kt*32+{0..7},  elems 8..15 = K kt*32+{16..23}
//   lanes 16-31 (grp=1), M=lane-16: elems 0..7 = K kt*32+{8..15}, elems 8..15 = K kt*32+{24..31}
__device__ __forceinline__ v16h load_afrag(const _Float16* P, int mrow, int kt, int grp) {
  const int k0 = kt * 32 + grp * 8;
  const int k1 = k0 + 16;
  half8 h0 = *(const half8*)(P + mrow * Lsz + k0);
  half8 h1 = *(const half8*)(P + mrow * Lsz + k1);
  v16h a;
#pragma unroll
  for (int e = 0; e < 8; ++e) { a[e] = h0[e]; a[8 + e] = h1[e]; }
  return a;
}

__global__ void __launch_bounds__(256)
asg_main_kernel(const float* __restrict__ inputs, const float* __restrict__ trans,
                const int* __restrict__ targets, const int* __restrict__ in_len,
                const int* __restrict__ tg_len, float* __restrict__ ws_full,
                float* __restrict__ ws_align, int T)
{
  // FCC cross-wave staging (blocks 0..3 only)
  __shared__ _Float16 Phi[16][Lsz];
  __shared__ _Float16 Plo[16][Lsz];
  __shared__ float    G[16][Lsz];

  const int tid = threadIdx.x;

  if (blockIdx.x < 4) {
    // ========== FCC: log-semiring scan as exp-domain WMMA GEMM, 2 barriers/step ==========
    const int b0   = blockIdx.x * 16;          // batches [b0, b0+16)
    const int wave = tid >> 5;
    const int lane = tid & 31;
    const int grp  = lane >> 4;
    const int nl   = lane & 15;
    const int m    = tid >> 4;                 // alpha row (one per 16-lane half-wave)
    const int q    = tid & 15;                 // column group within row
    const int j0   = q * 4;                    // 4 consecutive labels per thread
    const int mylen = in_len[b0 + m];

    // Persistent B fragments: B[k=j][n=i] = exp(trans[i][j]), split f16 hi/lo.
    v16h Bhi0 = {}, Bhi1 = {}, Blo0 = {}, Blo1 = {};
    if (wave < 4) {
      const int i = wave * 16 + nl;
#pragma unroll
      for (int e = 0; e < 16; ++e) {
        const int jk = grp * 8 + (e & 7) + (e >> 3) * 16;
        float v0 = expf(trans[i * Lsz + jk]);
        float v1 = expf(trans[i * Lsz + 32 + jk]);
        _Float16 h0 = (_Float16)v0, h1 = (_Float16)v1;
        Bhi0[e] = h0; Blo0[e] = (_Float16)(v0 - (float)h0);
        Bhi1[e] = h1; Blo1[e] = (_Float16)(v1 - (float)h1);
      }
    }

    // alpha lives in registers: 4 floats/thread, row m spread over its half-wave
    float4 av = *(const float4*)(inputs + (size_t)(b0 + m) * Lsz + j0);

    for (int t = 1; t < T; ++t) {
      const float4 xv = *(const float4*)(inputs + (((size_t)t * Bsz) + b0 + m) * Lsz + j0);

      // row max: barrier-free shuffle reduction within the 16-lane half-wave
      float mv = fmaxf(fmaxf(av.x, av.y), fmaxf(av.z, av.w));
#pragma unroll
      for (int off = 1; off < 16; off <<= 1)
        mv = fmaxf(mv, __shfl_xor(mv, off, 32));

      // P = exp(alpha - m), split into f16 hi + lo, staged to LDS for WMMA A-frags
      {
        float p0 = fast_exp(av.x - mv), p1 = fast_exp(av.y - mv);
        float p2 = fast_exp(av.z - mv), p3 = fast_exp(av.w - mv);
        half4 hv, lv;
        hv[0] = (_Float16)p0; lv[0] = (_Float16)(p0 - (float)hv[0]);
        hv[1] = (_Float16)p1; lv[1] = (_Float16)(p1 - (float)hv[1]);
        hv[2] = (_Float16)p2; lv[2] = (_Float16)(p2 - (float)hv[2]);
        hv[3] = (_Float16)p3; lv[3] = (_Float16)(p3 - (float)hv[3]);
        *(half4*)&Phi[m][j0] = hv;
        *(half4*)&Plo[m][j0] = lv;
      }
      __syncthreads();                          // P visible to WMMA waves

      // C(16x64) = P x exp(trans)^T via WMMA, split-f16 (~fp32 accuracy)
      if (wave < 4) {
        v16h ah0 = load_afrag(&Phi[0][0], nl, 0, grp);
        v16h ah1 = load_afrag(&Phi[0][0], nl, 1, grp);
        v16h al0 = load_afrag(&Plo[0][0], nl, 0, grp);
        v16h al1 = load_afrag(&Plo[0][0], nl, 1, grp);
        v8f c = {};
        c = __builtin_amdgcn_wmma_f32_16x16x32_f16(false, ah0, false, Bhi0, (short)0, c, false, false);
        c = __builtin_amdgcn_wmma_f32_16x16x32_f16(false, ah1, false, Bhi1, (short)0, c, false, false);
        c = __builtin_amdgcn_wmma_f32_16x16x32_f16(false, al0, false, Bhi0, (short)0, c, false, false);
        c = __builtin_amdgcn_wmma_f32_16x16x32_f16(false, al1, false, Bhi1, (short)0, c, false, false);
        c = __builtin_amdgcn_wmma_f32_16x16x32_f16(false, ah0, false, Blo0, (short)0, c, false, false);
        c = __builtin_amdgcn_wmma_f32_16x16x32_f16(false, ah1, false, Blo1, (short)0, c, false, false);
        const int iout = wave * 16 + nl;        // C layout: VGPR r -> M = 8*grp + r, N = lane&15
#pragma unroll
        for (int r = 0; r < 8; ++r) G[8 * grp + r][iout] = c[r];
      }
      __syncthreads();                          // G visible to all waves

      // alpha = x + m + log(C), gated on t < input_length
      if (t < mylen) {
        const float4 gv = *(const float4*)&G[m][j0];
        av.x = xv.x + mv + fast_log(gv.x);
        av.y = xv.y + mv + fast_log(gv.y);
        av.z = xv.z + mv + fast_log(gv.z);
        av.w = xv.w + mv + fast_log(gv.w);
      }
      // next iteration's Phi/Plo stores are separated from this step's WMMA reads
      // by the G barrier above; no end-of-step barrier needed.
    }

    // full[b] = logsumexp_j alpha[b, :] (shuffle reduction within half-wave)
    {
      float mx = fmaxf(fmaxf(av.x, av.y), fmaxf(av.z, av.w));
#pragma unroll
      for (int off = 1; off < 16; off <<= 1)
        mx = fmaxf(mx, __shfl_xor(mx, off, 32));
      float s = fast_exp(av.x - mx) + fast_exp(av.y - mx) +
                fast_exp(av.z - mx) + fast_exp(av.w - mx);
#pragma unroll
      for (int off = 1; off < 16; off <<= 1)
        s += __shfl_xor(s, off, 32);
      if (q == 0) ws_full[b0 + m] = mx + fast_log(s);
    }
  } else {
    // ========== FAC: barrier-free single-wave scan, one block per batch ==========
    const int b = blockIdx.x - 4;
    if (tid >= 32) return;                      // one wave32 does the whole chain
    const int lane = tid;                       // lane owns states s = lane*8 .. lane*8+7

    int   tg[8];
    float st[8], mv[8], a[8];
#pragma unroll
    for (int k = 0; k < 8; ++k) tg[k] = targets[b * Ssz + lane * 8 + k];
#pragma unroll
    for (int k = 0; k < 8; ++k) st[k] = trans[tg[k] * Lsz + tg[k]];
    {
      const int tprev0 = __shfl_up(tg[7], 1, 32);   // targets[s-1] across lane boundary
      mv[0] = (lane == 0) ? NEGV : trans[tg[0] * Lsz + tprev0];
#pragma unroll
      for (int k = 1; k < 8; ++k) mv[k] = trans[tg[k] * Lsz + tg[k - 1]];
    }
#pragma unroll
    for (int k = 0; k < 8; ++k) a[k] = NEGV;
    if (lane == 0) a[0] = inputs[(size_t)b * Lsz + tg[0]];   // emit[0, b, 0]

    const int ilen = in_len[b];
    for (int t = 1; t < T; ++t) {
      if (t < ilen) {                            // uniform over the wave
        const float* xr = inputs + (((size_t)t * Bsz) + b) * Lsz;
        const float shifted = __shfl_up(a[7], 1, 32);  // old alpha[s-1] at lane boundary
#pragma unroll
        for (int k = 7; k >= 0; --k) {           // descending: a[k-1] still old
          const float ap = (k == 0) ? shifted : a[k - 1];
          const float la = a[k] + st[k];
          const float lb = ap + mv[k];
          const float mm = fmaxf(la, lb);
          const float z  = fast_exp(fminf(la, lb) - mm);      // z in (0, 1]
          a[k] = xr[tg[k]] + mm + fast_log(1.f + z);          // log1p, arg in [1, 2]
        }
      }
    }

    // aligned[b] = alpha[b, target_len-1]
    {
      const int s = tg_len[b] - 1;
      const int kk = s & 7;
      float r = a[0];
#pragma unroll
      for (int k = 1; k < 8; ++k) r = (kk == k) ? a[k] : r;
      if (lane == (s >> 3)) ws_align[b] = r;
    }
  }
}

__global__ void __launch_bounds__(64)
asg_reduce_kernel(const float* __restrict__ ws_full, const float* __restrict__ ws_align,
                  float* __restrict__ out)
{
  __shared__ float red[Bsz];
  const int tid = threadIdx.x;
  red[tid] = ws_full[tid] - ws_align[tid];
  __syncthreads();
  for (int off = 32; off >= 1; off >>= 1) {
    if (tid < off) red[tid] += red[tid + off];
    __syncthreads();
  }
  if (tid == 0) out[0] = red[0] / (float)Bsz;
}

extern "C" void kernel_launch(void* const* d_in, const int* in_sizes, int n_in,
                              void* d_out, int out_size, void* d_ws, size_t ws_size,
                              hipStream_t stream) {
  const float* inputs  = (const float*)d_in[0];   // (T, B, L) f32
  const float* trans   = (const float*)d_in[1];   // (L, L) f32
  const int*   targets = (const int*)d_in[2];     // (B, S) i32
  const int*   in_len  = (const int*)d_in[3];     // (B,) i32
  const int*   tg_len  = (const int*)d_in[4];     // (B,) i32
  const int T = in_sizes[0] / (Bsz * Lsz);

  float* ws_full  = (float*)d_ws;        // 64 floats
  float* ws_align = ws_full + Bsz;       // 64 floats

  // blocks 0..3: FCC (16 batches each, WMMA scan); blocks 4..67: FAC (one wave each)
  asg_main_kernel<<<4 + Bsz, 256, 0, stream>>>(inputs, trans, targets, in_len, tg_len,
                                               ws_full, ws_align, T);
  asg_reduce_kernel<<<1, Bsz, 0, stream>>>(ws_full, ws_align, (float*)d_out);
}
